// Experts_7748121002269
// MI455X (gfx1250) — matrix operand order repaired and checked by reading, not compiled, not run on hardware
//
#include <hip/hip_runtime.h>

// Problem constants (from reference): B=2, E=8, C=8192, H=512, F=2048
#define B_ 2
#define E_ 8
#define C_ 8192
#define H_ 512
#define F_ 2048

typedef __attribute__((ext_vector_type(16))) __bf16 v16bf;
typedef __attribute__((ext_vector_type(8)))  __bf16 v8bf;
typedef __attribute__((ext_vector_type(8)))  float  v8f;
typedef __attribute__((ext_vector_type(4)))  float  v4f;

#define XS_STRIDE 520   // 512 + 8 bf16 pad -> 1040B row stride (16B aligned, bank-spread)
#define HS_STRIDE 136   // 128 + 8 bf16 pad -> 272B row stride

__device__ __forceinline__ v16bf cat8(v8bf a, v8bf b) {
  return __builtin_shufflevector(a, b, 0,1,2,3,4,5,6,7,8,9,10,11,12,13,14,15);
}

// tanh-approx GELU rewritten as x * sigmoid(2u): uses hw v_exp_f32 + v_rcp_f32
__device__ __forceinline__ float gelu_fast(float x) {
  float u2 = x * (1.5957691216057308f + 0.07135481627f * x * x);  // 2*u
  return x * __builtin_amdgcn_rcpf(1.0f + __expf(-u2));
}

// -------- Prep: fp32 [E][R][Cc] -> bf16 [E][Cc][R] (convert + transpose) --------
__global__ __launch_bounds__(256)
void transpose_cvt_bf16(const float* __restrict__ src, __bf16* __restrict__ dst,
                        int R, int Cc) {
  __shared__ __bf16 tile[64][65];
  const int tilesC = Cc >> 6;
  const int tilesR = R >> 6;
  const int tpe = tilesR * tilesC;
  const int e = blockIdx.x / tpe;
  const int t = blockIdx.x % tpe;
  const int r0 = (t / tilesC) << 6;
  const int c0 = (t % tilesC) << 6;
  const int tx = threadIdx.x & 63;
  const int ty = threadIdx.x >> 6;     // 0..3
  const float* s = src + (size_t)e * R * Cc;
  #pragma unroll
  for (int i = 0; i < 16; ++i) {
    int r = ty + i * 4;
    tile[r][tx] = (__bf16)s[(size_t)(r0 + r) * Cc + (c0 + tx)];
  }
  __syncthreads();
  __bf16* d = dst + (size_t)e * R * Cc;
  #pragma unroll
  for (int i = 0; i < 16; ++i) {
    int c = ty + i * 4;
    d[(size_t)(c0 + c) * R + (r0 + tx)] = tile[tx][c];
  }
}

// -------- Fused expert FFN: out = gelu(X @ w1 + b1) @ w2 + b2 --------
// grid: B*E*(C/64) blocks of 256 threads; each block -> 64 rows x H outputs.
// GEMM1: wave = 32x32 register tile (2x2 WMMA tiles) of the 64x128 h-chunk.
// GEMM2: wave = 64x64 register tile (4x4 WMMA tiles) of the 64x512 output.
__global__ __launch_bounds__(256, 1)
void experts_ffn(const float* __restrict__ X, const float* __restrict__ b1,
                 const float* __restrict__ b2, const __bf16* __restrict__ w1t,
                 const __bf16* __restrict__ w2t, float* __restrict__ out) {
  __shared__ __align__(16) __bf16 xs[64 * XS_STRIDE];  // X tile 64 x 512 (bf16)
  __shared__ __align__(16) __bf16 hs[64 * HS_STRIDE];  // gelu(h) chunk 64 x 128

  const int tid  = threadIdx.x;
  const int pair = blockIdx.x >> 7;     // (b*E + e); C/64 = 128 row-blocks per pair
  const int rb   = blockIdx.x & 127;
  const int e    = pair & (E_ - 1);
  const int row0 = rb * 64;

  // ---- Stage X tile: fp32 global -> bf16 LDS (coalesced float4 loads) ----
  const float* xsrc = X + ((size_t)pair * C_ + row0) * H_;
  #pragma unroll
  for (int i = 0; i < 32; ++i) {
    int lin = i * 1024 + tid * 4;       // 256 thr * 4 elems
    int r = lin >> 9;
    int c = lin & 511;
    v4f v = *(const v4f*)(xsrc + (size_t)r * H_ + c);
    __bf16* dp = &xs[r * XS_STRIDE + c];
    dp[0] = (__bf16)v.x; dp[1] = (__bf16)v.y; dp[2] = (__bf16)v.z; dp[3] = (__bf16)v.w;
  }
  __syncthreads();

  const int wave = tid >> 5;            // 0..7
  const int lane = tid & 31;
  const int l15  = lane & 15;
  const int sel  = lane >> 4;           // A: K-group {0..15} vs {16..31}
  const int mh   = wave >> 2;           // GEMM1 row half   (0..1) -> rows mh*32
  const int nh   = wave & 3;            // GEMM1 col quarter(0..3) -> cols nh*32

  v8f acc2[4][4] = {};                  // GEMM2: 64 rows x 64 cols per wave

  const __bf16* w1b  = w1t + ((size_t)e * F_ + nh * 32 + l15) * H_;   // lane = F-col
  const __bf16* w2b  = w2t + ((size_t)e * H_ + wave * 64 + l15) * F_; // lane = H-col
  const __bf16* xr0  = &xs[(mh * 32      + l15) * XS_STRIDE];
  const __bf16* xr1  = &xs[(mh * 32 + 16 + l15) * XS_STRIDE];

  for (int chunk = 0; chunk < 16; ++chunk) {  // F in chunks of 128
    const int fbase = chunk * 128;

    // ---- GEMM1: 32x32 tile of h = X @ w1 (K = 512) ----
    v8f acc1[2][2] = {};
    const __bf16* w1c = w1b + (size_t)fbase * H_;
    #pragma unroll
    for (int kk = 0; kk < 16; ++kk) {
      const int ab = kk * 32 + sel * 16;
      const int kb = kk * 32 + sel * 16;
      v16bf a0 = cat8(*(const v8bf*)(xr0 + ab), *(const v8bf*)(xr0 + ab + 8));
      v16bf a1 = cat8(*(const v8bf*)(xr1 + ab), *(const v8bf*)(xr1 + ab + 8));
      v16bf bf0 = *(const v16bf*)(w1c + kb);
      v16bf bf1 = *(const v16bf*)(w1c + (size_t)16 * H_ + kb);
      acc1[0][0] = __builtin_amdgcn_wmma_f32_16x16x32_bf16(false, a0, false, bf0, (short)0, acc1[0][0], false, false);
      acc1[0][1] = __builtin_amdgcn_wmma_f32_16x16x32_bf16(false, a0, false, bf1, (short)0, acc1[0][1], false, false);
      acc1[1][0] = __builtin_amdgcn_wmma_f32_16x16x32_bf16(false, a1, false, bf0, (short)0, acc1[1][0], false, false);
      acc1[1][1] = __builtin_amdgcn_wmma_f32_16x16x32_bf16(false, a1, false, bf1, (short)0, acc1[1][1], false, false);
    }

    __syncthreads();  // previous chunk's GEMM2 done reading hs

    // ---- bias + GELU, write bf16 h-chunk to LDS ----
    #pragma unroll
    for (int i = 0; i < 2; ++i) {
      #pragma unroll
      for (int j = 0; j < 2; ++j) {
        const int col  = nh * 32 + j * 16 + l15;
        const float bb = b1[e * F_ + fbase + col];
        #pragma unroll
        for (int r = 0; r < 8; ++r) {
          float v = acc1[i][j][r] + bb;           // D-layout: lane<16 -> M=r, else M=r+8
          int row = mh * 32 + i * 16 + (sel ? r + 8 : r);
          hs[row * HS_STRIDE + col] = (__bf16)gelu_fast(v);
        }
      }
    }
    __syncthreads();  // hs ready for all waves

    // ---- GEMM2 accumulate: 64x64 tile, out += gh @ w2 (K = 128) ----
    const __bf16* w2c = w2b + fbase;
    #pragma unroll
    for (int kk = 0; kk < 4; ++kk) {
      const int ab = kk * 32 + sel * 16;
      const int kb = kk * 32 + sel * 16;
      v16bf afr[4];
      #pragma unroll
      for (int mt = 0; mt < 4; ++mt) {
        const __bf16* hr = &hs[(mt * 16 + l15) * HS_STRIDE];
        afr[mt] = cat8(*(const v8bf*)(hr + ab), *(const v8bf*)(hr + ab + 8));
      }
      v16bf bfr[4];
      #pragma unroll
      for (int nt = 0; nt < 4; ++nt)
        bfr[nt] = *(const v16bf*)(w2c + (size_t)nt * 16 * F_ + kb);
      #pragma unroll
      for (int mt = 0; mt < 4; ++mt) {
        #pragma unroll
        for (int nt = 0; nt < 4; ++nt) {
          acc2[mt][nt] = __builtin_amdgcn_wmma_f32_16x16x32_bf16(
              false, afr[mt], false, bfr[nt], (short)0, acc2[mt][nt], false, false);
        }
      }
    }
  }

  // ---- Epilogue: + b2, coalesced fp32 stores ----
  float* obase = out + ((size_t)pair * C_ + row0) * H_;
  #pragma unroll
  for (int nt = 0; nt < 4; ++nt) {
    const int n    = wave * 64 + nt * 16 + l15;
    const float bb = b2[e * H_ + n];
    #pragma unroll
    for (int mt = 0; mt < 4; ++mt) {
      #pragma unroll
      for (int r = 0; r < 8; ++r) {
        int row = mt * 16 + (sel ? r + 8 : r);
        obase[(size_t)row * H_ + n] = acc2[mt][nt][r] + bb;
      }
    }
  }
}

extern "C" void kernel_launch(void* const* d_in, const int* in_sizes, int n_in,
                              void* d_out, int out_size, void* d_ws, size_t ws_size,
                              hipStream_t stream) {
  const float* X  = (const float*)d_in[0];   // [B,E,C,H]
  const float* w1 = (const float*)d_in[1];   // [E,H,F]
  const float* b1 = (const float*)d_in[2];   // [E,F]
  const float* w2 = (const float*)d_in[3];   // [E,F,H]
  const float* b2 = (const float*)d_in[4];   // [E,H]
  float* out = (float*)d_out;

  __bf16* w1t = (__bf16*)d_ws;                         // [E,F,H] bf16 (16 MB)
  __bf16* w2t = w1t + (size_t)E_ * F_ * H_;            // [E,H,F] bf16 (16 MB)

  // Prep: convert + transpose weights (weights fit in L2 afterwards)
  transpose_cvt_bf16<<<E_ * (H_ / 64) * (F_ / 64), 256, 0, stream>>>(w1, w1t, H_, F_);
  transpose_cvt_bf16<<<E_ * (F_ / 64) * (H_ / 64), 256, 0, stream>>>(w2, w2t, F_, H_);

  // Fused FFN
  experts_ffn<<<B_ * E_ * (C_ / 64), 256, 0, stream>>>(X, b1, b2, w1t, w2t, out);
}
